// Multihead_Attention_4105988735528
// MI455X (gfx1250) — matrix operand order repaired and checked by reading, not compiled, not run on hardware
//
#include <hip/hip_runtime.h>
#include <hip/hip_bf16.h>

typedef __attribute__((ext_vector_type(16))) __bf16 v16bf;
typedef __attribute__((ext_vector_type(2)))  __bf16 v2bf;
typedef __attribute__((ext_vector_type(8)))  float  v8f;

union FragU { unsigned u[8]; v16bf v; };

// Pack two fp32 -> one dword of 2x bf16 (single v_cvt_pk_bf16_f32).
__device__ __forceinline__ unsigned pack_bf16_pair(float lo, float hi) {
    v2bf p = { (__bf16)lo, (__bf16)hi };
    return __builtin_bit_cast(unsigned, p);
}

// ---------------------------------------------------------------------------
// WMMA fragment gathers from LDS tiles stored as [rows][cols] bf16, packed
// pairwise into u32 (cols even). cols_u = cols/2 (u32 stride per row).
//
// A-operand mapping (16x32, 16-bit; ISA 7.12.2):
//   lane(0-15/16-31) -> matrix row (row0 + lane&15)
//   vgpr t: k = half*8 + 2*(t&3) + (t>=4 ? 16 : 0), elements (k, k+1)
// ---------------------------------------------------------------------------
__device__ __forceinline__ v16bf ldsfrag_a(const unsigned* lds, int row0,
                                           int col0, int cols_u) {
    const int lane = threadIdx.x & 31;
    const int half = lane >> 4;
    const unsigned* p = lds + (row0 + (lane & 15)) * cols_u + (col0 >> 1);
    FragU f;
#pragma unroll
    for (int t = 0; t < 8; ++t) {
        const int k = half * 8 + ((t & 3) << 1) + ((t >= 4) ? 16 : 0);
        f.u[t] = p[k >> 1];
    }
    return f.v;
}

// ---------------------------------------------------------------------------
// B-operand mapping (32x16, 16-bit; ISA 7.12.5 pattern):
//   lane(0-15/16-31) -> matrix column (row of transposed-stored tile)
//   vgpr t: k = half*8 + 2*(t&3) + (t>=4 ? 16 : 0), elements (k, k+1)
// Tile must be stored transposed: [n][k].
// ---------------------------------------------------------------------------
__device__ __forceinline__ v16bf ldsfrag_b(const unsigned* lds, int row0,
                                           int col0, int cols_u) {
    const int lane = threadIdx.x & 31;
    const int half = lane >> 4;
    const unsigned* p = lds + (row0 + (lane & 15)) * cols_u + (col0 >> 1);
    FragU f;
#pragma unroll
    for (int t = 0; t < 8; ++t) {
        f.u[t] = p[half * 4 + (t & 3) + ((t >= 4) ? 8 : 0)];
    }
    return f.v;
}

// ===========================================================================
// GEMM with bias: C[M,N] = A[M,K] @ W[K,N] + bias[N]   (fp32 in/out, bf16 MACs)
// 128 threads (4 waves). Block tile 128(M) x 64(N), K-step 32.
// Wave w owns 32 M rows; B fragments shared across both row tiles.
// ===========================================================================
__global__ __launch_bounds__(128)
void gemm_bias_bf16w(const float* __restrict__ A, const float* __restrict__ W,
                     const float* __restrict__ bias, float* __restrict__ C,
                     int M, int N, int K) {
    __shared__ unsigned lds_a[128 * 16];  // [row 0..127][k 0..31] bf16-pairs
    __shared__ unsigned lds_wt[64 * 16];  // [n 0..63][k 0..31] (W transposed)

    const int tid  = threadIdx.x;
    const int wave = tid >> 5;
    const int lane = tid & 31;
    const int n0 = blockIdx.x * 64;
    const int m0 = blockIdx.y * 128;

    v8f acc0[4] = {};
    v8f acc1[4] = {};

    for (int kb = 0; kb < K; kb += 32) {
        __syncthreads();
        // Speculative prefetch of next K-step tiles (dropped if OOB).
        __builtin_prefetch(A + (size_t)(m0 + tid) * K + kb + 32, 0, 1);
        __builtin_prefetch(
            W + (size_t)(kb + 32 + (tid >> 2)) * N + n0 + (tid & 3) * 16, 0, 1);

        // A tile 128x32 -> bf16 pairs: 1024 float4 loads, 8 per thread
#pragma unroll
        for (int i = 0; i < 8; ++i) {
            const int p   = tid + i * 128;
            const int row = p >> 3;            // 8 quads per row
            const int q4  = (p & 7) << 2;
            const float4 v =
                *(const float4*)(A + (size_t)(m0 + row) * K + kb + q4);
            uint2 pk;
            pk.x = pack_bf16_pair(v.x, v.y);
            pk.y = pack_bf16_pair(v.z, v.w);
            *(uint2*)&lds_a[p * 2] = pk;       // row*16 + q4/2 == 2p
        }
        // W tile 32x64 -> transposed [n][k]: row-pair x n-quad, 2 per thread
#pragma unroll
        for (int i = 0; i < 2; ++i) {
            const int p  = tid + i * 128;      // 256 items
            const int kp = p >> 4;             // 0..15 (pair of k rows)
            const int n4 = (p & 15) << 2;      // 0,4,..,60
            const float* s0 = W + (size_t)(kb + kp * 2) * N + n0 + n4;
            const float4 r0 = *(const float4*)s0;
            const float4 r1 = *(const float4*)(s0 + N);
            lds_wt[(n4 + 0) * 16 + kp] = pack_bf16_pair(r0.x, r1.x);
            lds_wt[(n4 + 1) * 16 + kp] = pack_bf16_pair(r0.y, r1.y);
            lds_wt[(n4 + 2) * 16 + kp] = pack_bf16_pair(r0.z, r1.z);
            lds_wt[(n4 + 3) * 16 + kp] = pack_bf16_pair(r0.w, r1.w);
        }
        __syncthreads();

        const v16bf af0 = ldsfrag_a(lds_a, wave * 32, 0, 16);
        const v16bf af1 = ldsfrag_a(lds_a, wave * 32 + 16, 0, 16);
#pragma unroll
        for (int t = 0; t < 4; ++t) {
            const v16bf bf = ldsfrag_b(lds_wt, t * 16, 0, 16);
            acc0[t] = __builtin_amdgcn_wmma_f32_16x16x32_bf16(
                false, af0, false, bf, (short)0, acc0[t], false, false);
            acc1[t] = __builtin_amdgcn_wmma_f32_16x16x32_bf16(
                false, af1, false, bf, (short)0, acc1[t], false, false);
        }
    }

    // Epilogue: C/D layout row = r + 8*half, col = lane&15
    const int half = lane >> 4;
    const int col  = lane & 15;
#pragma unroll
    for (int t = 0; t < 4; ++t) {
        const int n = n0 + t * 16 + col;
        const float bv = bias[n];
#pragma unroll
        for (int r = 0; r < 8; ++r) {
            const int m = m0 + wave * 32 + r + half * 8;
            C[(size_t)m * N + n]        = acc0[t][r] + bv;
            C[(size_t)(m + 16) * N + n] = acc1[t][r] + bv;
        }
    }
}

// ===========================================================================
// Flash attention. qkv layout: [B, S, H*(3*64)] with per-head groups
// [q(64) | k(64) | v(64)]. One block = (b, h, 64 q rows); wave = 16 q rows.
// Online softmax over K/V strips of 32 rows. Output O: [B, S, H*64] fp32.
// ===========================================================================
#define D3 3072
#define DH 64

__global__ __launch_bounds__(128)
void flash_attn(const float* __restrict__ qkv, float* __restrict__ O, int S) {
    __shared__ unsigned lds_k[32 * 32];     // [kr 0..31][d 0..63] pairs
    __shared__ unsigned lds_vt[64 * 16];    // [d 0..63][kr 0..31] pairs (V^T)
    __shared__ unsigned lds_p[4][16 * 16];  // per-wave P: [qr 0..15][kr 0..31]

    const int tid  = threadIdx.x;
    const int wave = tid >> 5;
    const int lane = tid & 31;
    const int half = lane >> 4;

    const int qblk = blockIdx.x;       // S/64 q blocks
    const int bh   = blockIdx.y;       // B*H
    const int b    = bh >> 4;
    const int h    = bh & 15;

    const int q0 = qblk * 64 + wave * 16;

    // Q fragments (16 rows x 64 d) in A-layout, loaded once from global fp32.
    v16bf qf0, qf1;
    {
        const int qr = lane & 15;
        const float* qp = qkv + (size_t)(b * S + q0 + qr) * D3 + h * (3 * DH);
        FragU f0, f1;
#pragma unroll
        for (int t = 0; t < 8; ++t) {
            const int k = half * 8 + ((t & 3) << 1) + ((t >= 4) ? 16 : 0);
            f0.u[t] = pack_bf16_pair(qp[k],      qp[k + 1]);
            f1.u[t] = pack_bf16_pair(qp[32 + k], qp[32 + k + 1]);
        }
        qf0 = f0.v;
        qf1 = f1.v;
    }

    v8f o0 = {}, o1 = {}, o2 = {}, o3 = {};
    float mrow[8], lrow[8];
#pragma unroll
    for (int r = 0; r < 8; ++r) { mrow[r] = -3.0e38f; lrow[r] = 0.0f; }

    const float* kbase = qkv + (size_t)b * S * D3 + h * (3 * DH) + DH;
    const float* vbase = kbase + DH;

    for (int kb = 0; kb < S; kb += 32) {
        __syncthreads();
        // Speculative prefetch of next strip (dropped if OOB).
        __builtin_prefetch(
            kbase + (size_t)(kb + 32 + (tid >> 2)) * D3 + (tid & 3) * 16, 0, 1);
        __builtin_prefetch(
            vbase + (size_t)(kb + 32 + (tid >> 2)) * D3 + (tid & 3) * 16, 0, 1);

        // K strip 32x64 -> lds_k [kr][d]: 512 float4 loads, 4 per thread
#pragma unroll
        for (int i = 0; i < 4; ++i) {
            const int p  = tid + i * 128;
            const int kr = p >> 4;          // 16 quads per row
            const int q4 = (p & 15) << 2;
            const float4 v =
                *(const float4*)(kbase + (size_t)(kb + kr) * D3 + q4);
            uint2 pk;
            pk.x = pack_bf16_pair(v.x, v.y);
            pk.y = pack_bf16_pair(v.z, v.w);
            *(uint2*)&lds_k[p * 2] = pk;    // kr*32 + q4/2 == 2p
        }
        // V strip 32x64 -> transposed lds_vt [d][kr]: row-pair x d-quad
#pragma unroll
        for (int i = 0; i < 2; ++i) {
            const int p  = tid + i * 128;   // 256 items
            const int kp = p >> 4;          // 0..15 (pair of k rows)
            const int d4 = (p & 15) << 2;
            const float* s0 = vbase + (size_t)(kb + kp * 2) * D3 + d4;
            const float4 r0 = *(const float4*)s0;
            const float4 r1 = *(const float4*)(s0 + D3);
            lds_vt[(d4 + 0) * 16 + kp] = pack_bf16_pair(r0.x, r1.x);
            lds_vt[(d4 + 1) * 16 + kp] = pack_bf16_pair(r0.y, r1.y);
            lds_vt[(d4 + 2) * 16 + kp] = pack_bf16_pair(r0.z, r1.z);
            lds_vt[(d4 + 3) * 16 + kp] = pack_bf16_pair(r0.w, r1.w);
        }
        __syncthreads();

        // Scores: two 16x16 tiles (k rows 0..15 and 16..31), K-dim = 64
        v8f s0 = {}, s1 = {};
        {
            const v16bf b00 = ldsfrag_b(lds_k,  0,  0, 32);
            const v16bf b01 = ldsfrag_b(lds_k,  0, 32, 32);
            s0 = __builtin_amdgcn_wmma_f32_16x16x32_bf16(
                false, qf0, false, b00, (short)0, s0, false, false);
            s0 = __builtin_amdgcn_wmma_f32_16x16x32_bf16(
                false, qf1, false, b01, (short)0, s0, false, false);
            const v16bf b10 = ldsfrag_b(lds_k, 16,  0, 32);
            const v16bf b11 = ldsfrag_b(lds_k, 16, 32, 32);
            s1 = __builtin_amdgcn_wmma_f32_16x16x32_bf16(
                false, qf0, false, b10, (short)0, s1, false, false);
            s1 = __builtin_amdgcn_wmma_f32_16x16x32_bf16(
                false, qf1, false, b11, (short)0, s1, false, false);
        }

        // Online softmax (scale = 1/sqrt(64) = 0.125). Row r of a vgpr lives
        // in a fixed 16-lane half; xor masks 1..8 never cross the boundary.
        {
            __bf16* pst = (__bf16*)lds_p[wave];
            const int col = lane & 15;
#pragma unroll
            for (int r = 0; r < 8; ++r) {
                const float a = s0[r] * 0.125f;
                const float c = s1[r] * 0.125f;
                float bm = fmaxf(a, c);
#pragma unroll
                for (int off = 1; off < 16; off <<= 1)
                    bm = fmaxf(bm, __shfl_xor(bm, off, 32));
                const float mn    = fmaxf(mrow[r], bm);
                const float alpha = __expf(mrow[r] - mn);
                mrow[r] = mn;
                const float p0 = __expf(a - mn);
                const float p1 = __expf(c - mn);
                float rs = p0 + p1;
#pragma unroll
                for (int off = 1; off < 16; off <<= 1)
                    rs += __shfl_xor(rs, off, 32);
                lrow[r] = lrow[r] * alpha + rs;
                o0[r] *= alpha; o1[r] *= alpha; o2[r] *= alpha; o3[r] *= alpha;
                const int prow = r + half * 8;
                pst[prow * 32 + col]      = (__bf16)p0;
                pst[prow * 32 + 16 + col] = (__bf16)p1;
            }
        }
        // Wave-local C-layout -> A-layout transpose through LDS: fence DS ops.
        asm volatile("s_wait_dscnt 0" ::: "memory");

        // P @ V : A = P (16x32), B = V^T tiles (32x16 each over d blocks)
        {
            const v16bf pf = ldsfrag_a(lds_p[wave], 0, 0, 16);
            const v16bf v0 = ldsfrag_b(lds_vt,  0, 0, 16);
            const v16bf v1 = ldsfrag_b(lds_vt, 16, 0, 16);
            const v16bf v2 = ldsfrag_b(lds_vt, 32, 0, 16);
            const v16bf v3 = ldsfrag_b(lds_vt, 48, 0, 16);
            o0 = __builtin_amdgcn_wmma_f32_16x16x32_bf16(
                false, pf, false, v0, (short)0, o0, false, false);
            o1 = __builtin_amdgcn_wmma_f32_16x16x32_bf16(
                false, pf, false, v1, (short)0, o1, false, false);
            o2 = __builtin_amdgcn_wmma_f32_16x16x32_bf16(
                false, pf, false, v2, (short)0, o2, false, false);
            o3 = __builtin_amdgcn_wmma_f32_16x16x32_bf16(
                false, pf, false, v3, (short)0, o3, false, false);
        }
    }

    // Finalize: O[b, q, h*64 + d] = o / l
    const int col = lane & 15;
#pragma unroll
    for (int r = 0; r < 8; ++r) {
        const float inv = 1.0f / lrow[r];
        const int q = q0 + r + half * 8;
        float* dst = O + (size_t)(b * S + q) * 1024 + h * DH;
        dst[col]      = o0[r] * inv;
        dst[16 + col] = o1[r] * inv;
        dst[32 + col] = o2[r] * inv;
        dst[48 + col] = o3[r] * inv;
    }
}

// ===========================================================================
extern "C" void kernel_launch(void* const* d_in, const int* in_sizes, int n_in,
                              void* d_out, int out_size, void* d_ws, size_t ws_size,
                              hipStream_t stream) {
    const float* x    = (const float*)d_in[0];
    const float* Wqkv = (const float*)d_in[1];
    const float* bqkv = (const float*)d_in[2];
    const float* Wo   = (const float*)d_in[3];
    const float* bo   = (const float*)d_in[4];
    float* out = (float*)d_out;

    const int B = 2, S = 2048, D = 1024, H = 16;
    const int M = B * S;                    // 4096

    float* qkv = (float*)d_ws;              // [M, 3D]  = 48 MB
    float* Ov  = qkv + (size_t)M * 3 * D;   // [M, D]   = 16 MB

    dim3 blk(128);

    // 1) qkv = x @ Wqkv + bqkv
    gemm_bias_bf16w<<<dim3((3 * D) / 64, M / 128), blk, 0, stream>>>(
        x, Wqkv, bqkv, qkv, M, 3 * D, D);

    // 2) flash attention over heads
    flash_attn<<<dim3(S / 64, B * H), blk, 0, stream>>>(qkv, Ov, S);

    // 3) out = Ov @ Wo + bo
    gemm_bias_bf16w<<<dim3(D / 64, M / 128), blk, 0, stream>>>(
        Ov, Wo, bo, out, M, D, D);
}
